// DynamicLinkPredictor_37941741093472
// MI455X (gfx1250) — compile-verified, matrix operand order checked by reading, not run on hardware
//
#include <hip/hip_runtime.h>

typedef __attribute__((ext_vector_type(16))) _Float16 v16h;
typedef __attribute__((ext_vector_type(8)))  _Float16 v8h;
typedef __attribute__((ext_vector_type(8)))  float    v8f;

#define N_NODES 512
#define N_BATCH 2
#define N_EDGES 16384
#define HID     64

// load a 16-half WMMA fragment: two aligned 16B chunks at +kb and +kb+16
static __device__ __forceinline__ v16h frag16(const _Float16* p) {
  v8h lo = *(const v8h*)p;
  v8h hi = *(const v8h*)(p + 16);
  return __builtin_shufflevector(lo, hi, 0, 1, 2, 3, 4, 5, 6, 7,
                                         8, 9, 10, 11, 12, 13, 14, 15);
}

// ============================ graph preprocessing ============================
__global__ void k_degrees(const int* __restrict__ row, const int* __restrict__ col,
                          const float* __restrict__ w,
                          float* dout, float* din, float* cnt) {
  int e = blockIdx.x * blockDim.x + threadIdx.x;
  if (e >= N_EDGES) return;
  atomicAdd(&dout[row[e]], w[e]);
  atomicAdd(&din[col[e]],  w[e]);
  atomicAdd(&cnt[col[e]],  1.0f);
}

__global__ void k_dinv(const float* cnt, float* dinv) {
  int n = blockIdx.x * blockDim.x + threadIdx.x;
  if (n < N_NODES) dinv[n] = rsqrtf(cnt[n] + 2.0f);
}

__global__ void k_edge_norms(const int* __restrict__ row, const int* __restrict__ col,
                             const float* dout, const float* din, const float* dinv,
                             float* no, float* ni, float* ng) {
  int e = blockIdx.x * blockDim.x + threadIdx.x;
  if (e >= N_EDGES) return;
  float d0 = dout[row[e]];
  float d1 = din[col[e]];
  no[e] = d0 > 0.f ? 1.f / d0 : 0.f;
  ni[e] = d1 > 0.f ? 1.f / d1 : 0.f;
  ng[e] = dinv[row[e]] * dinv[col[e]];
}

// ============================ initial node embedding ============================
__global__ void k_init_h(const float* __restrict__ x,
                         const float* fp_w, const float* fp_b, const float* fp_g, const float* fp_be,
                         const float* tp_w, const float* tp_b, const float* tp_g, const float* tp_be,
                         const float* node_emb, float* __restrict__ h) {
  __shared__ float sh[HID];
  int bn = blockIdx.x;
  int n  = bn & (N_NODES - 1);
  int j  = threadIdx.x;
  const float* xr = x + bn * 60;                 // x[b,n,:,:] flattened (F*T)
  float s1 = fp_b[j];
  #pragma unroll
  for (int f = 0; f < 6; ++f) s1 += xr[f * 10 + 9] * fp_w[f * HID + j];
  sh[j] = s1; __syncthreads();
  float mu = 0.f, m2 = 0.f;
  for (int k = 0; k < HID; ++k) { float t = sh[k]; mu += t; m2 += t * t; }
  mu *= (1.f / HID); m2 = m2 * (1.f / HID) - mu * mu;
  float h1 = (s1 - mu) * rsqrtf(m2 + 1e-5f) * fp_g[j] + fp_be[j];
  h1 = h1 > 0.f ? h1 : 0.f;
  __syncthreads();
  float s2 = tp_b[j];
  for (int k = 0; k < 60; ++k) s2 += xr[k] * tp_w[k * HID + j];
  sh[j] = s2; __syncthreads();
  mu = 0.f; m2 = 0.f;
  for (int k = 0; k < HID; ++k) { float t = sh[k]; mu += t; m2 += t * t; }
  mu *= (1.f / HID); m2 = m2 * (1.f / HID) - mu * mu;
  float h2 = (s2 - mu) * rsqrtf(m2 + 1e-5f) * tp_g[j] + tp_be[j];
  h2 = h2 > 0.f ? h2 : 0.f;
  h[bn * HID + j] = h1 + h2 + node_emb[n * HID + j];
}

// ============================ Chebyshev scatter propagation ============================
__global__ void k_prop(const float* __restrict__ src_o, const float* __restrict__ src_i,
                       const int* __restrict__ row, const int* __restrict__ col,
                       const float* __restrict__ no, const float* __restrict__ ni,
                       float* __restrict__ dst_o, float* __restrict__ dst_i) {
  int idx = blockIdx.x * blockDim.x + threadIdx.x;
  if (idx >= N_BATCH * N_EDGES * HID) return;
  int c = idx & 63;
  int e = (idx >> 6) & (N_EDGES - 1);
  int b = idx >> 20;
  int rs = (b * N_NODES + row[e]) * HID + c;
  int ds = (b * N_NODES + col[e]) * HID + c;
  atomicAdd(&dst_o[ds], src_o[rs] * no[e]);
  atomicAdd(&dst_i[ds], src_i[rs] * ni[e]);
}

// Tx2 = 2*prop(Tx1) - X  (Chebyshev recurrence, k=2)
__global__ void k_cheb2(float* P2o, float* P2i, const float* __restrict__ h) {
  int i = blockIdx.x * blockDim.x + threadIdx.x;
  if (i >= N_BATCH * N_NODES * HID) return;
  P2o[i] = 2.f * P2o[i] - h[i];
  P2i[i] = 2.f * P2i[i] - h[i];
}

// pack stacked operand U = [X | P1o | P1i | P2o | P2i]  (1024 x 320, f16)
__global__ void k_pack_u(const float* h, const float* P1o, const float* P1i,
                         const float* P2o, const float* P2i, _Float16* __restrict__ U) {
  int idx = blockIdx.x * blockDim.x + threadIdx.x;
  if (idx >= N_BATCH * N_NODES * HID) return;
  int r = idx >> 6, c = idx & 63;
  _Float16* u = U + r * 320;
  u[c]       = (_Float16)h[idx];
  u[64 + c]  = (_Float16)P1o[idx];
  u[128 + c] = (_Float16)P1i[idx];
  u[192 + c] = (_Float16)P2o[idx];
  u[256 + c] = (_Float16)P2i[idx];
}

// pack gate weights TRANSPOSED: out is N x K = 128 x 320 f16.
// logical rows (K): [Wg(0,0)+Wg(1,0) | Wg(0,1) | Wg(1,1) | Wg(0,2) | Wg(1,2)] (top-H slices),
// logical cols (N): [z-gate | h-gate].  dc_w layout: (2,3,128,64)
__global__ void k_pack_wgate(const float* __restrict__ Wz, const float* __restrict__ Wh,
                             _Float16* __restrict__ WzhT) {
  int idx = blockIdx.x * blockDim.x + threadIdx.x;
  if (idx >= 320 * 128) return;
  int r = idx >> 7;          // K index 0..319
  int nn = idx & 127;        // N index 0..127
  const float* W = (nn < HID) ? Wz : Wh;
  int c = nn & (HID - 1);
  int seg = r >> 6;
  int rr = r & 63;
  float v;
  if (seg == 0)      v = W[((0 * 3 + 0) * 128 + rr) * 64 + c] + W[((1 * 3 + 0) * 128 + rr) * 64 + c];
  else if (seg == 1) v = W[((0 * 3 + 1) * 128 + rr) * 64 + c];
  else if (seg == 2) v = W[((1 * 3 + 1) * 128 + rr) * 64 + c];
  else if (seg == 3) v = W[((0 * 3 + 2) * 128 + rr) * 64 + c];
  else               v = W[((1 * 3 + 2) * 128 + rr) * 64 + c];
  WzhT[nn * 320 + r] = (_Float16)v;
}

// generic f32 (K x N) -> f16 transposed (N x K)
__global__ void k_pack_t(const float* __restrict__ a, _Float16* __restrict__ o, int K, int N) {
  int idx = blockIdx.x * blockDim.x + threadIdx.x;
  if (idx >= K * N) return;
  int k = idx / N, n = idx - k * N;
  o[n * K + k] = (_Float16)a[idx];
}

// ============================ generic f16 WMMA GEMM ============================
// C(f32, MxN) = A(f16, MxK row-major) @ B, with B given TRANSPOSED (N x K row-major).
// One wave per 16x16 output tile; both fragments load as 2x16B contiguous chunks
// (CDNA5 ISA 7.12.2 layout: halves 0-7 -> K+kb..K+kb+7, halves 8-15 -> K+kb+16..K+kb+23).
__global__ void __launch_bounds__(128)
k_wmma_gemm(const _Float16* __restrict__ A, const _Float16* __restrict__ Bt,
            float* __restrict__ C, int M, int K, int N) {
  int wave = (blockIdx.x * blockDim.x + threadIdx.x) >> 5;
  int lane = threadIdx.x & 31;
  int ntiles = N >> 4;
  int mt = wave / ntiles;
  int nt = wave - mt * ntiles;
  if (mt >= (M >> 4)) return;                 // wave-uniform: EXEC stays all-ones
  int lm = lane & 15;
  int kb = (lane < 16) ? 0 : 8;
  const _Float16* Arow = A  + ((mt << 4) + lm) * K + kb;
  const _Float16* Brow = Bt + ((nt << 4) + lm) * K + kb;
  v8f acc = {0.f, 0.f, 0.f, 0.f, 0.f, 0.f, 0.f, 0.f};
  for (int k0 = 0; k0 < K; k0 += 32) {
    v16h a = frag16(Arow + k0);
    v16h b = frag16(Brow + k0);
    acc = __builtin_amdgcn_wmma_f32_16x16x32_f16(false, a, false, b, (short)0, acc, false, false);
  }
  int mbase = (mt << 4) + ((lane < 16) ? 0 : 8);
  int ncol  = (nt << 4) + lm;
  #pragma unroll
  for (int rr = 0; rr < 8; ++rr)
    C[(mbase + rr) * N + ncol] = acc[rr];
}

// ============================ GRU gates (r-gate is dead code: skipped) ============================
__global__ void k_gate(const float* __restrict__ G, const float* bz, const float* bh,
                       float* __restrict__ htmp) {
  int idx = blockIdx.x * blockDim.x + threadIdx.x;
  if (idx >= N_BATCH * N_NODES * HID) return;
  int r = idx >> 6, j = idx & 63;
  float z = 1.f / (1.f + expf(-(G[r * 128 + j] + bz[j])));
  float t = tanhf(G[r * 128 + 64 + j] + bh[j]);
  htmp[idx] = (1.f - z) * t;          // h_temp -> also the "hiddens" output
}

__global__ void k_f32_to_f16(const float* __restrict__ a, _Float16* __restrict__ o, int n) {
  int i = blockIdx.x * blockDim.x + threadIdx.x;
  if (i < n) o[i] = (_Float16)a[i];
}

__global__ void k_gcn_scatter(const float* __restrict__ xw,
                              const int* __restrict__ row, const int* __restrict__ col,
                              const float* __restrict__ ng, float* __restrict__ hsp) {
  int idx = blockIdx.x * blockDim.x + threadIdx.x;
  if (idx >= N_BATCH * N_EDGES * HID) return;
  int c = idx & 63;
  int e = (idx >> 6) & (N_EDGES - 1);
  int b = idx >> 20;
  atomicAdd(&hsp[(b * N_NODES + col[e]) * HID + c],
            xw[(b * N_NODES + row[e]) * HID + c] * ng[e]);
}

__global__ void k_finish(const float* __restrict__ htmp, const float* __restrict__ hsp,
                         const float* __restrict__ xw, const float* __restrict__ dinv,
                         const float* gcn_b, const float* g, const float* be,
                         float* __restrict__ h, float* __restrict__ hprev) {
  __shared__ float sh[HID];
  int bn = blockIdx.x;
  int n = bn & (N_NODES - 1);
  int j = threadIdx.x;
  int idx = bn * HID + j;
  float di = dinv[n];
  float v = htmp[idx] + hsp[idx] + xw[idx] * (2.f * di * di) + gcn_b[j];
  sh[j] = v; __syncthreads();
  float mu = 0.f, m2 = 0.f;
  for (int k = 0; k < HID; ++k) { float t = sh[k]; mu += t; m2 += t * t; }
  mu *= (1.f / HID); m2 = m2 * (1.f / HID) - mu * mu;
  float hh = (v - mu) * rsqrtf(m2 + 1e-5f) * g[j] + be[j];
  hh = hh > 0.f ? hh : 0.f;
  float hold = h[idx];
  hprev[idx] = hold;
  h[idx] = hh + hold;
}

// ============================ head factor packing ============================
__global__ void k_pack_hf(const float* h, const float* hprev, _Float16* __restrict__ hf16) {
  int idx = blockIdx.x * blockDim.x + threadIdx.x;
  if (idx >= N_BATCH * N_NODES * 128) return;
  int r = idx >> 7, c = idx & 127;
  hf16[idx] = (_Float16)((c < HID) ? h[r * HID + c] : hprev[r * HID + (c - HID)]);
}

// W1 is (4H,2H)=(256,128). Pack TRANSPOSED to N x K = 256 x 128:
// logical (k,n): n<128 -> W1_top[k,n], n>=128 -> W1_bot[k,n-128]
__global__ void k_pack_w1(const float* __restrict__ W1, _Float16* __restrict__ W1pT) {
  int idx = blockIdx.x * blockDim.x + threadIdx.x;
  if (idx >= 128 * 256) return;
  int k = idx >> 8, nn = idx & 255;
  float v = (nn < 128) ? W1[k * 128 + nn] : W1[(128 + k) * 128 + (nn - 128)];
  W1pT[nn * 128 + k] = (_Float16)v;
}

// ============================ fused link-prediction head ============================
// tile: 16 i's x 8 j's = 128 pairs.  A = relu(LN(Afac_i + Bfac_j + b1)) staged f16 in LDS,
// then 128x128x64 WMMA GEMM vs W2^T (LDS), D written over the A region, per-row LN+relu+dot(w3).
__global__ void __launch_bounds__(128)
k_head(const float* __restrict__ Cfac, const _Float16* __restrict__ W2pT,
       const float* b1, const float* b2,
       const float* g1, const float* be1, const float* g2, const float* be2,
       const float* w3, const float* b3, float* __restrict__ logits) {
  __shared__ float smAD[8192];                 // 32KB: A (128x128 f16) aliased with D (128x64 f32)
  __shared__ _Float16 W2l[64 * 128];           // 16KB, N x K (transposed)
  _Float16* Ash = (_Float16*)smAD;
  float* Dsh = smAD;
  int bz = blockIdx.z, it = blockIdx.y, jt = blockIdx.x;
  int tid = threadIdx.x;                        // 128 threads = 4 waves
  for (int t = tid; t < 64 * 128; t += 128) W2l[t] = W2pT[t];

  int r = tid;                                  // row 0..127 of this tile
  int ii = r >> 3, jj = r & 7;
  int irow = bz * N_NODES + it * 16 + ii;
  int jrow = bz * N_NODES + jt * 8 + jj;
  const float* Ca = Cfac + irow * 256;          // Afac (cols 0..127)
  const float* Cb = Cfac + jrow * 256 + 128;    // Bfac (cols 128..255)
  float mu = 0.f, m2 = 0.f;
  for (int c = 0; c < 128; ++c) {
    float t = Ca[c] + Cb[c] + b1[c];
    mu += t; m2 += t * t;
  }
  mu *= (1.f / 128.f); m2 = m2 * (1.f / 128.f) - mu * mu;
  float rs = rsqrtf(m2 + 1e-5f);
  for (int c = 0; c < 128; ++c) {
    float t = ((Ca[c] + Cb[c] + b1[c]) - mu) * rs * g1[c] + be1[c];
    Ash[r * 128 + c] = (_Float16)(t > 0.f ? t : 0.f);
  }
  __syncthreads();

  int wave = tid >> 5, lane = tid & 31;
  int lm = lane & 15;
  int kb = (lane < 16) ? 0 : 8;
  const v8f vz = {0.f, 0.f, 0.f, 0.f, 0.f, 0.f, 0.f, 0.f};
  v8f acc[2][4];
  #pragma unroll
  for (int mi = 0; mi < 2; ++mi)
    #pragma unroll
    for (int nt = 0; nt < 4; ++nt) acc[mi][nt] = vz;

  #pragma unroll
  for (int mi = 0; mi < 2; ++mi) {
    const _Float16* Arow = Ash + ((wave * 2 + mi) * 16 + lm) * 128 + kb;
    for (int k0 = 0; k0 < 128; k0 += 32) {
      v16h a = frag16(Arow + k0);
      #pragma unroll
      for (int nt = 0; nt < 4; ++nt) {
        v16h bf = frag16(W2l + (nt * 16 + lm) * 128 + kb + k0);
        acc[mi][nt] = __builtin_amdgcn_wmma_f32_16x16x32_f16(
            false, a, false, bf, (short)0, acc[mi][nt], false, false);
      }
    }
  }
  __syncthreads();                              // all Ash reads done; reuse LDS for D
  #pragma unroll
  for (int mi = 0; mi < 2; ++mi) {
    int mbase = (wave * 2 + mi) * 16 + ((lane < 16) ? 0 : 8);
    #pragma unroll
    for (int nt = 0; nt < 4; ++nt) {
      int nc = nt * 16 + lm;
      #pragma unroll
      for (int rr = 0; rr < 8; ++rr)
        Dsh[(mbase + rr) * 64 + nc] = acc[mi][nt][rr];
    }
  }
  __syncthreads();

  mu = 0.f; m2 = 0.f;
  for (int c = 0; c < HID; ++c) {
    float t = Dsh[r * 64 + c] + b2[c];
    mu += t; m2 += t * t;
  }
  mu *= (1.f / HID); m2 = m2 * (1.f / HID) - mu * mu;
  rs = rsqrtf(m2 + 1e-5f);
  float dot = 0.f;
  for (int c = 0; c < HID; ++c) {
    float t = (Dsh[r * 64 + c] + b2[c] - mu) * rs * g2[c] + be2[c];
    dot += (t > 0.f ? t : 0.f) * w3[c];
  }
  logits[irow * N_NODES + jt * 8 + jj] = dot + b3[0];
}

// ============================ host ============================
extern "C" void kernel_launch(void* const* d_in, const int* in_sizes, int n_in,
                              void* d_out, int out_size, void* d_ws, size_t ws_size,
                              hipStream_t stream) {
  (void)in_sizes; (void)n_in; (void)out_size; (void)ws_size;
  const float* x  = (const float*)d_in[0];
  const int*   ei = (const int*)d_in[1];
  const int*   row = ei;
  const int*   col = ei + N_EDGES;
  const float* ew = (const float*)d_in[2];
  // params flattened in sorted-key (JAX pytree) order, starting at index 3
  const float* dc_bh[2] = {(const float*)d_in[3],  (const float*)d_in[9]};
  const float* dc_bz[2] = {(const float*)d_in[5],  (const float*)d_in[11]};
  const float* dc_wh[2] = {(const float*)d_in[6],  (const float*)d_in[12]};
  const float* dc_wz[2] = {(const float*)d_in[8],  (const float*)d_in[14]};
  const float* fp_b  = (const float*)d_in[15];
  const float* fp_be = (const float*)d_in[16];
  const float* fp_g  = (const float*)d_in[17];
  const float* fp_w  = (const float*)d_in[18];
  const float* gcn_b[2] = {(const float*)d_in[19], (const float*)d_in[21]};
  const float* gcn_w[2] = {(const float*)d_in[20], (const float*)d_in[22]};
  const float* ln_b[2]  = {(const float*)d_in[23], (const float*)d_in[25]};
  const float* ln_g[2]  = {(const float*)d_in[24], (const float*)d_in[26]};
  const float* lp_b1  = (const float*)d_in[27];
  const float* lp_b2  = (const float*)d_in[28];
  const float* lp_b3  = (const float*)d_in[29];
  const float* lp_be1 = (const float*)d_in[30];
  const float* lp_be2 = (const float*)d_in[31];
  const float* lp_g1  = (const float*)d_in[32];
  const float* lp_g2  = (const float*)d_in[33];
  const float* lp_w1  = (const float*)d_in[34];
  const float* lp_w2  = (const float*)d_in[35];
  const float* lp_w3  = (const float*)d_in[36];
  const float* node_emb = (const float*)d_in[37];
  const float* tp_b  = (const float*)d_in[38];
  const float* tp_be = (const float*)d_in[39];
  const float* tp_g  = (const float*)d_in[40];
  const float* tp_w  = (const float*)d_in[41];

  // ---- workspace carving (256B aligned) ----
  char* wsb = (char*)d_ws;
  size_t off = 0;
  auto carve = [&](size_t bytes) -> char* {
    char* p = wsb + off;
    off += (bytes + 255) & ~(size_t)255;
    return p;
  };
  const int NBH = N_BATCH * N_NODES * HID;        // 65536
  float*    degO = (float*)carve(N_NODES * 4);
  float*    degI = (float*)carve(N_NODES * 4);
  float*    cnt  = (float*)carve(N_NODES * 4);
  float*    dinv = (float*)carve(N_NODES * 4);
  float*    nrmO = (float*)carve(N_EDGES * 4);
  float*    nrmI = (float*)carve(N_EDGES * 4);
  float*    nrmG = (float*)carve(N_EDGES * 4);
  float*    h    = (float*)carve(NBH * 4);
  float*    hprev= (float*)carve(NBH * 4);
  float*    P1o  = (float*)carve(NBH * 4);        // P1o..P2i contiguous (sizes are 256B multiples)
  float*    P1i  = (float*)carve(NBH * 4);
  float*    P2o  = (float*)carve(NBH * 4);
  float*    P2i  = (float*)carve(NBH * 4);
  _Float16* U    = (_Float16*)carve(1024 * 320 * 2);
  _Float16* WzhT = (_Float16*)carve(320 * 128 * 2);
  float*    G    = (float*)carve(1024 * 128 * 4);
  _Float16* h16  = (_Float16*)carve(NBH * 2);
  _Float16* gw16 = (_Float16*)carve(HID * HID * 2);
  float*    xw   = (float*)carve(NBH * 4);
  float*    hsp  = (float*)carve(NBH * 4);
  _Float16* hf16 = (_Float16*)carve(1024 * 128 * 2);
  _Float16* W1pT = (_Float16*)carve(128 * 256 * 2);
  float*    Cfac = (float*)carve(1024 * 256 * 4);
  _Float16* W2pT = (_Float16*)carve(128 * 64 * 2);

  float* logits = (float*)d_out;                              // B*N*N
  float* hidden_base = (float*)d_out + N_BATCH * N_NODES * N_NODES;

  // ---- graph preprocessing ----
  hipMemsetAsync(degO, 0, 3 * N_NODES * 4, stream);           // degO,degI,cnt contiguous
  k_degrees<<<(N_EDGES + 255) / 256, 256, 0, stream>>>(row, col, ew, degO, degI, cnt);
  k_dinv<<<2, 256, 0, stream>>>(cnt, dinv);
  k_edge_norms<<<(N_EDGES + 255) / 256, 256, 0, stream>>>(row, col, degO, degI, dinv, nrmO, nrmI, nrmG);

  // ---- initial embedding ----
  k_init_h<<<N_BATCH * N_NODES, HID, 0, stream>>>(x, fp_w, fp_b, fp_g, fp_be,
                                                  tp_w, tp_b, tp_g, tp_be, node_emb, h);

  const int SCAT = N_BATCH * N_EDGES * HID;                   // 2M scatter threads
  for (int l = 0; l < 2; ++l) {
    // Chebyshev propagations (shared by z- and h-gates; dead r-gate skipped)
    hipMemsetAsync(P1o, 0, (size_t)4 * NBH * 4, stream);      // zero P1o..P2i
    k_prop<<<SCAT / 256, 256, 0, stream>>>(h, h, row, col, nrmO, nrmI, P1o, P1i);
    k_prop<<<SCAT / 256, 256, 0, stream>>>(P1o, P1i, row, col, nrmO, nrmI, P2o, P2i);
    k_cheb2<<<(NBH + 255) / 256, 256, 0, stream>>>(P2o, P2i, h);

    // fused gate GEMM: U(1024x320) @ Wzh(320x128) -> G   (B passed transposed)
    k_pack_u<<<(NBH + 255) / 256, 256, 0, stream>>>(h, P1o, P1i, P2o, P2i, U);
    k_pack_wgate<<<(320 * 128 + 255) / 256, 256, 0, stream>>>(dc_wz[l], dc_wh[l], WzhT);
    k_wmma_gemm<<<(64 * 8) / 4, 128, 0, stream>>>(U, WzhT, G, 1024, 320, 128);
    float* htmp = hidden_base + l * NBH;                      // hiddens[l] output slot
    k_gate<<<(NBH + 255) / 256, 256, 0, stream>>>(G, dc_bz[l], dc_bh[l], htmp);

    // GCN branch: xw = h @ gcn_w, scatter with symmetric norm, + self term
    k_f32_to_f16<<<(NBH + 255) / 256, 256, 0, stream>>>(h, h16, NBH);
    k_pack_t<<<(HID * HID + 255) / 256, 256, 0, stream>>>(gcn_w[l], gw16, HID, HID);
    k_wmma_gemm<<<(64 * 4) / 4, 128, 0, stream>>>(h16, gw16, xw, 1024, 64, 64);
    hipMemsetAsync(hsp, 0, (size_t)NBH * 4, stream);
    k_gcn_scatter<<<SCAT / 256, 256, 0, stream>>>(xw, row, col, nrmG, hsp);
    k_finish<<<N_BATCH * N_NODES, HID, 0, stream>>>(htmp, hsp, xw, dinv,
                                                    gcn_b[l], ln_g[l], ln_b[l], h, hprev);
  }

  // ---- head factors: Cfac = [h|hprev](1024x128) @ [W1_top|W1_bot](128x256) ----
  k_pack_hf<<<(1024 * 128 + 255) / 256, 256, 0, stream>>>(h, hprev, hf16);
  k_pack_w1<<<(128 * 256 + 255) / 256, 256, 0, stream>>>(lp_w1, W1pT);
  k_wmma_gemm<<<(64 * 16) / 4, 128, 0, stream>>>(hf16, W1pT, Cfac, 1024, 128, 256);
  k_pack_t<<<(128 * 64 + 255) / 256, 256, 0, stream>>>(lp_w2, W2pT, 128, 64);

  // ---- fused head over all (b,i,j) pairs ----
  dim3 hgrid(N_NODES / 8, N_NODES / 16, N_BATCH);             // 64 x 32 x 2
  k_head<<<hgrid, 128, 0, stream>>>(Cfac, W2pT, lp_b1, lp_b2, lp_g1, lp_be1,
                                    lp_g2, lp_be2, lp_w3, lp_b3, logits);
}
